// MyMultiHeadSelfAttention_6193342841624
// MI455X (gfx1250) — matrix operand order repaired and checked by reading, not compiled, not run on hardware
//
#include <hip/hip_runtime.h>

// MI455X (gfx1250) multi-head self-attention, wave32 + WMMA bf16 + async LDS copies.
// B=8, S=1024, D=1024, H=16, DH=64.
//
// Pipeline (all launches serial on `stream`):
//   0) cvt_f32_to_bf16: Wq/Wk/Wv/Wo -> bf16 (ws), and each activation -> bf16 (ws, reused)
//   1) gemm_wmma<QKV>: Q=(x@Wq^T+bq)*0.125, K, V -> bf16 (B,H,S,DH) in ws
//      double-buffered K-loop, tiles staged with global_load_async_to_lds_b128 (ASYNCcnt)
//   2) flash_attn: fused scores/softmax/context per (b,h,128-q-block) -> bf16 ctx (B,S,D)
//   3) gemm_wmma<f32 out>: out = ctx@Wo^T + bo -> f32 d_out

typedef __attribute__((ext_vector_type(8)))  float  v8f;
typedef __attribute__((ext_vector_type(16))) __bf16 v16bf;
typedef __attribute__((ext_vector_type(8)))  __bf16 v8bf;

#define BATCH 8
#define SEQ   1024
#define DIM   1024
#define HEADS 16
#define DH    64

static __device__ inline v16bf combine16(v8bf lo, v8bf hi) {
  v16bf r;
#pragma unroll
  for (int i = 0; i < 8; ++i) { r[i] = lo[i]; r[i + 8] = hi[i]; }
  return r;
}

static __device__ inline v8bf ld8(const __bf16* p) { return *(const v8bf*)p; }

// Async global->LDS copy of 16 bytes (8 bf16). Generic pointers into LDS carry the
// LDS byte offset in their low 32 bits (aperture mapping), which is what the
// instruction's LDS-address VGPR wants. Tracked by ASYNCcnt.
static __device__ inline void async_copy_b128(__bf16* lds_dst, const __bf16* gsrc) {
  unsigned lds_off = (unsigned)(size_t)lds_dst;
  asm volatile("global_load_async_to_lds_b128 %0, %1, off"
               :: "v"(lds_off), "v"(gsrc)
               : "memory");
}
static __device__ inline void wait_async0() {
  asm volatile("s_wait_asynccnt 0x0" ::: "memory");
}

// Reduce across the 16 lanes of a half-wave (xor masks 1,2,4,8 stay inside a half).
static __device__ inline float redmax16(float x) {
  x = fmaxf(x, __shfl_xor(x, 1, 32));
  x = fmaxf(x, __shfl_xor(x, 2, 32));
  x = fmaxf(x, __shfl_xor(x, 4, 32));
  x = fmaxf(x, __shfl_xor(x, 8, 32));
  return x;
}
static __device__ inline float redsum16(float x) {
  x += __shfl_xor(x, 1, 32);
  x += __shfl_xor(x, 2, 32);
  x += __shfl_xor(x, 4, 32);
  x += __shfl_xor(x, 8, 32);
  return x;
}

static __device__ inline v8f wmma_bf16(v16bf a, v16bf b, v8f c) {
  return __builtin_amdgcn_wmma_f32_16x16x32_bf16(
      /*neg_a=*/false, a, /*neg_b=*/false, b,
      /*c_mod=*/(short)0, c, /*reuse_a=*/false, /*reuse_b=*/false);
}

// ---------------------------------------------------------------------------
// Streaming f32 -> bf16 convert (8 elements/thread, 16B stores).
// ---------------------------------------------------------------------------
__global__ __launch_bounds__(256) void cvt_f32_to_bf16(const float* __restrict__ s,
                                                       __bf16* __restrict__ d) {
  const size_t i = ((size_t)blockIdx.x * 256 + threadIdx.x) * 8;
  float4 a = *(const float4*)(s + i);
  float4 b = *(const float4*)(s + i + 4);
  v8bf o;
  o[0] = (__bf16)a.x; o[1] = (__bf16)a.y; o[2] = (__bf16)a.z; o[3] = (__bf16)a.w;
  o[4] = (__bf16)b.x; o[5] = (__bf16)b.y; o[6] = (__bf16)b.z; o[7] = (__bf16)b.w;
  *(v8bf*)(d + i) = o;
}

// ---------------------------------------------------------------------------
// GEMM: out[M,N] = alpha * (A[M,K] @ W^T + bias), M=8192, N=K=1024, all-bf16 in.
// W is row-major (N,K) bf16, so B[k][n] = W[n*K+k] (contiguous along k).
// Block = 256 threads (8 waves) -> 128x128 tile; wave -> 32x64 sub-tile.
// Double-buffered LDS, staged with async global->LDS copies.
// ---------------------------------------------------------------------------
template <bool OUT_QKV>
__global__ __launch_bounds__(256) void gemm_wmma(const __bf16* __restrict__ A,
                                                 const __bf16* __restrict__ W,
                                                 const float* __restrict__ bias,
                                                 void* __restrict__ out,
                                                 float alpha) {
  __shared__ __bf16 As[2][128][32];  // [buf][m][k]  2 x 8 KiB
  __shared__ __bf16 Bs[2][128][32];  // [buf][n][k]  2 x 8 KiB

  const int tid  = threadIdx.x;
  const int wave = tid >> 5;
  const int lane = tid & 31;
  const int hl   = lane >> 4;   // half-wave
  const int l16  = lane & 15;

  const int m_blk = blockIdx.x * 128;
  const int n_blk = blockIdx.y * 128;
  const int wm = (wave & 3) * 32;   // 2 M-tiles
  const int wn = (wave >> 2) * 64;  // 4 N-tiles

  const int srow = tid >> 1;          // 0..127
  const int skc  = (tid & 1) * 16;    // 0 or 16

  v8f acc[2][4];
#pragma unroll
  for (int i = 0; i < 2; ++i)
#pragma unroll
    for (int j = 0; j < 4; ++j) acc[i][j] = (v8f)0.0f;

  const __bf16* aRow = A + (size_t)(m_blk + srow) * DIM + skc;
  const __bf16* wRow = W + (size_t)(n_blk + srow) * DIM + skc;

  // Prologue: stage tile kb=0 into buffer 0.
  async_copy_b128(&As[0][srow][skc],     aRow);
  async_copy_b128(&As[0][srow][skc + 8], aRow + 8);
  async_copy_b128(&Bs[0][srow][skc],     wRow);
  async_copy_b128(&Bs[0][srow][skc + 8], wRow + 8);
  wait_async0();
  __syncthreads();

  int buf = 0;
  for (int kb = 0; kb < DIM; kb += 32) {
    // Kick off the next tile into the other buffer while we compute.
    if (kb + 32 < DIM) {
      const int nb = buf ^ 1;
      async_copy_b128(&As[nb][srow][skc],     aRow + kb + 32);
      async_copy_b128(&As[nb][srow][skc + 8], aRow + kb + 40);
      async_copy_b128(&Bs[nb][srow][skc],     wRow + kb + 32);
      async_copy_b128(&Bs[nb][srow][skc + 8], wRow + kb + 40);
    }

    // A-fragment: lane row M=l16; elems 0..7 = k[hl*8 ..], 8..15 = k[hl*8+16 ..]
    v16bf af[2];
#pragma unroll
    for (int i = 0; i < 2; ++i) {
      const __bf16* r = &As[buf][wm + i * 16 + l16][0];
      af[i] = combine16(ld8(r + hl * 8), ld8(r + hl * 8 + 16));
    }
    // B-fragment: lane col N=l16; elems 0..7 = k[hl*8 ..], 8..15 = k[hl*8+16 ..]
    v16bf bfr[4];
#pragma unroll
    for (int j = 0; j < 4; ++j) {
      const __bf16* r = &Bs[buf][wn + j * 16 + l16][hl * 8];
      bfr[j] = combine16(ld8(r), ld8(r + 16));
    }
#pragma unroll
    for (int i = 0; i < 2; ++i)
#pragma unroll
      for (int j = 0; j < 4; ++j) acc[i][j] = wmma_bf16(af[i], bfr[j], acc[i][j]);

    wait_async0();
    __syncthreads();
    buf ^= 1;
  }

  // Epilogue: C layout -> row m = hl*8 + r (within 16-tile), col n = l16.
#pragma unroll
  for (int j = 0; j < 4; ++j) {
    const int col = n_blk + wn + j * 16 + l16;
    const float bv = bias[col];
#pragma unroll
    for (int i = 0; i < 2; ++i) {
#pragma unroll
      for (int r = 0; r < 8; ++r) {
        const int m = m_blk + wm + i * 16 + hl * 8 + r;
        const float v = alpha * (acc[i][j][r] + bv);
        if (OUT_QKV) {
          // (B,S,D) row m=(b,s), col=(h,d) -> bf16 (B,H,S,DH)
          const int b = m >> 10, s = m & 1023, h = col >> 6, d = col & 63;
          ((__bf16*)out)[(((size_t)(b * HEADS + h)) * SEQ + s) * DH + d] = (__bf16)v;
        } else {
          ((float*)out)[(size_t)m * DIM + col] = v;
        }
      }
    }
  }
}

// ---------------------------------------------------------------------------
// Flash attention: one block per (b, h, 128-row q block). 8 waves, each owns
// a 16-row q tile. Online softmax; K tile async-staged, V transposed via DS.
// ---------------------------------------------------------------------------
__global__ __launch_bounds__(256) void flash_attn(const void* __restrict__ Qv,
                                                  const void* __restrict__ Kv,
                                                  const void* __restrict__ Vv,
                                                  const int* __restrict__ mask,
                                                  const unsigned char* __restrict__ miss,
                                                  void* __restrict__ ctx_out) {
  __shared__ __bf16 Ks[32][64];       // [key][d]
  __shared__ __bf16 Vt[64][32];       // [d][key] (transposed)
  __shared__ __bf16 Ps[8][16][32];    // per-wave P relayout scratch

  const int b = blockIdx.z, h = blockIdx.y;
  const int qb = blockIdx.x * 128;
  const int tid  = threadIdx.x;
  const int wave = tid >> 5;
  const int lane = tid & 31;
  const int hl   = lane >> 4;
  const int l16  = lane & 15;

  const size_t head_off = (size_t)(b * HEADS + h) * SEQ * DH;
  const __bf16* Q = (const __bf16*)Qv + head_off;
  const __bf16* K = (const __bf16*)Kv + head_off;
  const __bf16* V = (const __bf16*)Vv + head_off;

  // Q A-fragments for this wave's 16 rows (K-dim = DH=64 -> 2 frags of 32).
  const int qrow = qb + wave * 16 + l16;
  v16bf qf[2];
#pragma unroll
  for (int s = 0; s < 2; ++s) {
    const __bf16* r = Q + (size_t)qrow * DH + s * 32 + hl * 8;
    qf[s] = combine16(ld8(r), ld8(r + 16));
  }

  float mrow[8], lrow[8];
  v8f cacc[4];
#pragma unroll
  for (int r = 0; r < 8; ++r) { mrow[r] = -1e30f; lrow[r] = 0.0f; }
#pragma unroll
  for (int j = 0; j < 4; ++j) cacc[j] = (v8f)0.0f;

  const int skey = tid >> 3;          // 0..31
  const int sd0  = (tid & 7) * 8;     // 0..56

  for (int kb = 0; kb < SEQ; kb += 32) {
    __syncthreads();
    // K tile: async global->LDS (rows are contiguous bf16).
    async_copy_b128(&Ks[skey][sd0], K + (size_t)(kb + skey) * DH + sd0);
    // V tile: load + transpose through registers into LDS.
    {
      v8bf vv = ld8(V + (size_t)(kb + skey) * DH + sd0);
#pragma unroll
      for (int i = 0; i < 8; ++i) Vt[sd0 + i][skey] = vv[i];
    }
    wait_async0();
    __syncthreads();

    // Scores: two 16x16 tiles (key cols), contract over d in 2 steps of 32.
    v8f sc[2];
#pragma unroll
    for (int t = 0; t < 2; ++t) {
      sc[t] = (v8f)0.0f;
#pragma unroll
      for (int s = 0; s < 2; ++s) {
        const __bf16* r = &Ks[t * 16 + l16][s * 32 + hl * 8];
        v16bf kf = combine16(ld8(r), ld8(r + 16));
        sc[t] = wmma_bf16(qf[s], kf, sc[t]);
      }
      // Padding mask: column key = kb + 16t + l16 for every element in lane.
      const int mv = mask[b * SEQ + kb + t * 16 + l16];
      if (mv == 0) {
#pragma unroll
        for (int r = 0; r < 8; ++r) sc[t][r] = -1e30f;
      }
    }

    // Online softmax per row (row = hl*8 + r; cols spread over 16 lanes of a half).
    float p0[8], p1[8];
#pragma unroll
    for (int r = 0; r < 8; ++r) {
      float tmax = redmax16(fmaxf(sc[0][r], sc[1][r]));
      float mnew = fmaxf(mrow[r], tmax);
      float rescale = __expf(mrow[r] - mnew);
      p0[r] = __expf(sc[0][r] - mnew);
      p1[r] = __expf(sc[1][r] - mnew);
      float rs = redsum16(p0[r] + p1[r]);
      lrow[r] = lrow[r] * rescale + rs;
      mrow[r] = mnew;
#pragma unroll
      for (int j = 0; j < 4; ++j) cacc[j][r] *= rescale;
    }

    // Relayout P (C layout) -> A layout via per-wave LDS scratch.
#pragma unroll
    for (int r = 0; r < 8; ++r) {
      Ps[wave][hl * 8 + r][l16]      = (__bf16)p0[r];
      Ps[wave][hl * 8 + r][16 + l16] = (__bf16)p1[r];
    }
    {
      const __bf16* pr = &Ps[wave][l16][0];
      v16bf pf = combine16(ld8(pr + hl * 8), ld8(pr + hl * 8 + 16));
#pragma unroll
      for (int j = 0; j < 4; ++j) {
        const __bf16* r = &Vt[j * 16 + l16][hl * 8];
        v16bf vf = combine16(ld8(r), ld8(r + 16));
        cacc[j] = wmma_bf16(pf, vf, cacc[j]);
      }
    }
  }

  // Normalize, head-drop, store ctx as bf16 (B,S,D) with col = h*64 + d.
  const bool missH = miss[h] != 0;
  __bf16* C = (__bf16*)ctx_out;
#pragma unroll
  for (int j = 0; j < 4; ++j) {
    const int d = j * 16 + l16;
#pragma unroll
    for (int r = 0; r < 8; ++r) {
      const int srow = qb + wave * 16 + hl * 8 + r;
      float v = cacc[j][r] / lrow[r];
      if (missH) v = 0.0f;
      C[((size_t)b * SEQ + srow) * DIM + h * DH + d] = (__bf16)v;
    }
  }
}

// ---------------------------------------------------------------------------
extern "C" void kernel_launch(void* const* d_in, const int* in_sizes, int n_in,
                              void* d_out, int out_size, void* d_ws, size_t ws_size,
                              hipStream_t stream) {
  const float* query = (const float*)d_in[0];
  const float* key   = (const float*)d_in[1];
  const float* value = (const float*)d_in[2];
  const int*   mask  = (const int*)d_in[3];
  const unsigned char* miss = (const unsigned char*)d_in[4];
  const float* Wq = (const float*)d_in[5];
  const float* bq = (const float*)d_in[6];
  const float* Wk = (const float*)d_in[7];
  const float* bk = (const float*)d_in[8];
  const float* Wv = (const float*)d_in[9];
  const float* bv = (const float*)d_in[10];
  const float* Wo = (const float*)d_in[11];
  const float* bo = (const float*)d_in[12];

  // Workspace layout (bf16 buffers), ~88 MiB total:
  //   Abf (reused per projection) | Q | K | V | ctx | Wq | Wk | Wv | Wo
  const size_t act = (size_t)BATCH * SEQ * DIM;   // 8 Mi elements
  const size_t wel = (size_t)DIM * DIM;           // 1 Mi elements
  char* ws = (char*)d_ws;
  __bf16* Abf = (__bf16*)(ws);
  __bf16* Qb  = (__bf16*)(ws + act * 2);
  __bf16* Kb  = (__bf16*)(ws + act * 4);
  __bf16* Vb  = (__bf16*)(ws + act * 6);
  __bf16* Cb  = (__bf16*)(ws + act * 8);
  __bf16* Wqb = (__bf16*)(ws + act * 10);
  __bf16* Wkb = (__bf16*)(ws + act * 10 + wel * 2);
  __bf16* Wvb = (__bf16*)(ws + act * 10 + wel * 4);
  __bf16* Wob = (__bf16*)(ws + act * 10 + wel * 6);

  const int actBlocks = (int)(act / 2048);  // 8 elems/thread, 256 thr/block
  const int wBlocks   = (int)(wel / 2048);
  dim3 ggrid(BATCH * SEQ / 128, DIM / 128);  // (64, 8)
  const float qscale = 0.125f;               // 1/sqrt(DH)

  // Weights -> bf16 once.
  cvt_f32_to_bf16<<<wBlocks, 256, 0, stream>>>(Wq, Wqb);
  cvt_f32_to_bf16<<<wBlocks, 256, 0, stream>>>(Wk, Wkb);
  cvt_f32_to_bf16<<<wBlocks, 256, 0, stream>>>(Wv, Wvb);
  cvt_f32_to_bf16<<<wBlocks, 256, 0, stream>>>(Wo, Wob);

  // Q/K/V projections (activation buffer reused serially).
  cvt_f32_to_bf16<<<actBlocks, 256, 0, stream>>>(query, Abf);
  gemm_wmma<true><<<ggrid, 256, 0, stream>>>(Abf, Wqb, bq, Qb, qscale);
  cvt_f32_to_bf16<<<actBlocks, 256, 0, stream>>>(key, Abf);
  gemm_wmma<true><<<ggrid, 256, 0, stream>>>(Abf, Wkb, bk, Kb, 1.0f);
  cvt_f32_to_bf16<<<actBlocks, 256, 0, stream>>>(value, Abf);
  gemm_wmma<true><<<ggrid, 256, 0, stream>>>(Abf, Wvb, bv, Vb, 1.0f);

  flash_attn<<<dim3(SEQ / 128, HEADS, BATCH), 256, 0, stream>>>(Qb, Kb, Vb, mask, miss, Cb);

  gemm_wmma<false><<<ggrid, 256, 0, stream>>>(Cb, Wob, bo, d_out, 1.0f);
}
